// Decoder_27358941676162
// MI455X (gfx1250) — matrix operand order repaired and checked by reading, not compiled, not run on hardware
//
#include <hip/hip_runtime.h>
#include <hip/hip_bf16.h>
#include <cmath>

// ---------------------------------------------------------------------------
// Types for CDNA5 WMMA (wave32): A/B = 16 bf16 per lane (8 VGPRs),
// C/D = 8 f32 per lane (8 VGPRs).
// ---------------------------------------------------------------------------
typedef __attribute__((ext_vector_type(16))) __bf16 v16bf;
typedef __attribute__((ext_vector_type(8)))  float  v8f;
typedef __attribute__((ext_vector_type(4)))  unsigned int u32x4;
typedef __attribute__((ext_vector_type(8)))  int  i32x8;
typedef __attribute__((ext_vector_type(4)))  int  i32x4;

constexpr int cT   = 2048;  // sequence length
constexpr int cH   = 12;    // heads
constexpr int cD   = 768;   // head dim (quirk: = EMB)
constexpr int cE   = 768;   // embedding dim
constexpr int cQKV = 9216;  // H * D
constexpr int cFF  = 3072;  // 4 * EMB

// Block tile 256(M) x 128(N), 8 waves in 4x2 grid, wave tile 64x64, K-step 32.
constexpr int BM = 256;
constexpr int BN = 128;

// Padded LDS strides (halves). 40*2=80B and 136*2=272B are 16B-aligned rows.
constexpr int SA = 40;   // A tile row stride (32 data + 8 pad halves)
constexpr int SB = 136;  // B tile row stride (128 data + 8 pad halves)

enum EpMode { EP_QKV = 0, EP_KT, EP_F32SCALE, EP_PV, EP_F32_BIAS_RES, EP_BF16_BIAS_GELU };

// ---------------------------------------------------------------------------
// Tensor Data Mover (async tile DMA global -> LDS), guarded by __has_builtin.
// ---------------------------------------------------------------------------
#if defined(__has_builtin)
#if __has_builtin(__builtin_amdgcn_tensor_load_to_lds) && __has_builtin(__builtin_amdgcn_s_wait_tensorcnt)
#define USE_TDM 1
#endif
#endif
#ifndef USE_TDM
#define USE_TDM 0
#endif

#if USE_TDM
// 2D tile load: tile_x elems (bf16) per row, tile_y rows, row stride in elems.
// pad_int/pad_amt: D# pad codes producing the padded LDS row stride.
__device__ __forceinline__ void tdm_load_2d(unsigned lds_addr, const void* gptr,
                                            unsigned tile_x, unsigned tile_y,
                                            unsigned stride,
                                            unsigned pad_int, unsigned pad_amt)
{
  unsigned long long ga = (unsigned long long)(size_t)gptr;
  u32x4 g0 = {};
  g0[0] = 1u;                                            // count=1 (valid), user mode
  g0[1] = lds_addr;                                      // lds_addr [63:32]
  g0[2] = (unsigned)ga;                                  // global_addr[31:0]
  g0[3] = (unsigned)((ga >> 32) & 0x01FFFFFFu) | (2u << 30);  // addr[56:32] | type=2

  unsigned td0 = stride;          // tensor_dim0 >= tile span (no OOB clip)
  unsigned td1 = 0x40000000u;     // tensor_dim1: huge (no OOB clip)
  i32x8 g1 = {};
  g1[0] = (int)((1u << 16)                  // data_size = 2 bytes
              | (1u << 20)                  // pad_enable
              | (pad_int << 22)             // pad_interval code
              | (pad_amt << 25));           // pad_amount code
  g1[1] = (int)((td0 & 0xFFFFu) << 16);                      // tensor_dim0[15:0]
  g1[2] = (int)((td0 >> 16) | ((td1 & 0xFFFFu) << 16));      // td0[31:16] | td1[15:0]
  g1[3] = (int)((td1 >> 16) | (tile_x << 16));               // td1[31:16] | tile_dim0
  g1[4] = (int)(tile_y & 0xFFFFu);                           // tile_dim1 (tile_dim2=0)
  g1[5] = (int)stride;                                       // tensor_dim0_stride[31:0]
  g1[6] = 0;                                                 // stride[47:32]=0, dim1_stride lo=0
  g1[7] = 0;

  i32x4 g2 = {};
  i32x4 g3 = {};
#if __clang_major__ >= 23
  i32x8 g4 = {};
  __builtin_amdgcn_tensor_load_to_lds(g0, g1, g2, g3, g4, 0);
#else
  __builtin_amdgcn_tensor_load_to_lds(g0, g1, g2, g3, 0);
#endif
}
#endif

__device__ __forceinline__ void tdm_wait()
{
#if USE_TDM
  __builtin_amdgcn_s_wait_tensorcnt((short)0);
#endif
}

// ---------------------------------------------------------------------------
// LDS fragment loads (ds_load_b128 pairs)
// A fragment (16x32, M x K): lane<16 -> row=lane, cols {c..c+7, c+16..c+23};
//                            lane>=16 -> row=lane-16, cols shifted by 8.
// ---------------------------------------------------------------------------
__device__ __forceinline__ v16bf lds_a_frag(const __bf16* p) {
  uint4 lo = *(const uint4*)(p);
  uint4 hi = *(const uint4*)(p + 16);
  v16bf f;
  __builtin_memcpy(&f, &lo, 16);
  __builtin_memcpy((char*)&f + 16, &hi, 16);
  return f;
}
// B fragment (32x16, K x N): lane -> column n, rows {r..r+7, r+16..r+23} (r = 8*(lane/16)).
__device__ __forceinline__ v16bf lds_b_frag(const __bf16* p) {
  v16bf f;
#pragma unroll
  for (int i = 0; i < 8; ++i) {
    f[i] = p[i * SB];
    f[8 + i] = p[(16 + i) * SB];
  }
  return f;
}

// ---------------------------------------------------------------------------
// bf16 WMMA GEMM with TDM-staged, double-buffered LDS tiles.
// C[M,N] = A[M,K] @ B[K,N]  (A,B row-major, lda=K, ldb=N)
// Block tile 256x128, 8 waves (4 along M x 2 along N), wave tile 64x64
// (4x4 WMMA tiles => 16 v_wmma per K-step of 32).
// M multiple of 256, N multiple of 128, K multiple of 32 (true for all GEMMs).
// ---------------------------------------------------------------------------
template<int EP>
__global__ __launch_bounds__(256)
void gemm_wmma_bf16(const __bf16* __restrict__ A, const __bf16* __restrict__ B,
                    void* __restrict__ C,
                    int M, int N, int K,
                    long sAz, long sBz, long sCz, int ldc,
                    const float* __restrict__ bias,
                    const float* __restrict__ res,
                    float scale, int causal)
{
  __shared__ __bf16 shA[2][BM * SA];  // 40,960 B
  __shared__ __bf16 shB[2][32 * SB];  // 17,408 B

  const int m0 = blockIdx.y * BM;
  const int n0 = blockIdx.x * BN;
  const int z  = blockIdx.z;

  // Causal skip: block fully above the diagonal -> softmax never reads it.
  if constexpr (EP == EP_F32SCALE) {
    if (causal && n0 >= m0 + BM) return;   // uniform per block, before any barrier
  }

  const __bf16* Az = A + (long)z * sAz;
  const __bf16* Bz = B + (long)z * sBz;

  const int tid  = threadIdx.x;
  const int wid  = tid >> 5;
  const int lane = tid & 31;
  const int wm   = (wid & 3) * 64;        // wave row offset in block tile
  const int wn   = (wid >> 2) * 64;       // wave col offset in block tile
  const int rlo  = lane & 15;
  const int hi2  = lane >> 4;             // 0/1

  // ---- tile staging: TDM descriptors (wave0: A, wave4: B) or vector copies ----
  auto stage = [&](int buf, int k0) {
#if USE_TDM
    if (wid == 0)
      tdm_load_2d((unsigned)(size_t)&shA[buf][0], Az + (long)m0 * K + k0,
                  /*tile_x=*/32, /*tile_y=*/BM, /*stride=*/(unsigned)K,
                  /*pad_int: 16 dwords*/3, /*pad_amt: 4 dwords*/3);
    else if (wid == 4)
      tdm_load_2d((unsigned)(size_t)&shB[buf][0], Bz + (long)k0 * N + n0,
                  /*tile_x=*/128, /*tile_y=*/32, /*stride=*/(unsigned)N,
                  /*pad_int: 64 dwords*/5, /*pad_amt: 4 dwords*/3);
#else
    // A tile: 256 rows x 32 halves = 1024 x 16B chunks
    for (int c = tid; c < BM * 4; c += 256) {
      int r = c >> 2, seg = c & 3;
      *(uint4*)&shA[buf][r * SA + seg * 8] =
          *(const uint4*)(Az + (long)(m0 + r) * K + k0 + seg * 8);
    }
    // B tile: 32 rows x 128 halves = 512 x 16B chunks
    for (int c = tid; c < 512; c += 256) {
      int r = c >> 4, seg = c & 15;
      *(uint4*)&shB[buf][r * SB + seg * 8] =
          *(const uint4*)(Bz + (long)(k0 + r) * N + n0 + seg * 8);
    }
#endif
  };

  v8f acc[4][4] = {};

  const int nsteps = K / 32;
  stage(0, 0);
  tdm_wait();
  __syncthreads();

  for (int s = 0; s < nsteps; ++s) {
    const int cur = s & 1;
    if (s + 1 < nsteps) stage(cur ^ 1, (s + 1) * 32);  // overlap DMA with compute

    v16bf bfr[4];
#pragma unroll
    for (int j = 0; j < 4; ++j)
      bfr[j] = lds_b_frag(&shB[cur][hi2 * 8 * SB + wn + j * 16 + rlo]);

#pragma unroll
    for (int i = 0; i < 4; ++i) {
      v16bf af = lds_a_frag(&shA[cur][(wm + i * 16 + rlo) * SA + hi2 * 8]);
#pragma unroll
      for (int j = 0; j < 4; ++j)
        acc[i][j] = __builtin_amdgcn_wmma_f32_16x16x32_bf16(
            false, af, false, bfr[j], (short)0, acc[i][j], false, false);
    }

    tdm_wait();          // next-buffer DMA landed
    __syncthreads();     // everyone done reading cur before it is overwritten
  }

  // Epilogue. C/D layout: vgpr g, lane l -> m = g + 8*(l/16), n = l%16.
  // 16x16 tiles never straddle a batch (2048) or head (768) boundary, so all
  // div/mod index math hoists to once per tile.
#pragma unroll
  for (int i = 0; i < 4; ++i) {
#pragma unroll
    for (int j = 0; j < 4; ++j) {
      const int mt = m0 + wm + i * 16;   // tile row base
      const int nt = n0 + wn + j * 16;   // tile col base
      if constexpr (EP == EP_QKV) {
        // out[b][h][t][d], m = b*T + t, n = h*D + d
        const int bb = mt / cT, t0 = mt - bb * cT;
        const int hh = nt / cD, d0 = nt - hh * cD;
        __bf16* O = (__bf16*)C;
        long base = ((long)(bb * cH + hh) * cT + t0 + hi2 * 8) * cD + d0 + rlo;
#pragma unroll
        for (int g = 0; g < 8; ++g)
          O[base + (long)g * cD] = (__bf16)acc[i][j][g];
      } else if constexpr (EP == EP_KT) {
        // out[b][h][d][t]: 8 contiguous t per lane -> one b128 store
        const int bb = mt / cT, t0 = mt - bb * cT;
        const int hh = nt / cD, d0 = nt - hh * cD;
        __bf16* O = (__bf16*)C;
        long base = ((long)(bb * cH + hh) * cD + d0 + rlo) * cT + t0 + hi2 * 8;
        __bf16 tmp[8];
#pragma unroll
        for (int g = 0; g < 8; ++g) tmp[g] = (__bf16)acc[i][j][g];
        uint4 pk;
        __builtin_memcpy(&pk, tmp, 16);
        *(uint4*)(O + base) = pk;
      } else if constexpr (EP == EP_F32SCALE) {
        float* O = (float*)C;
        long base = (long)z * sCz + (long)(mt + hi2 * 8) * ldc + nt + rlo;
#pragma unroll
        for (int g = 0; g < 8; ++g)
          O[base + (long)g * ldc] = acc[i][j][g] * scale;
      } else if constexpr (EP == EP_PV) {
        // out[b][t][h*D + n], m = t (per-head GEMM), z = b*H + h
        const int bb = z / cH, hh = z - bb * cH;
        __bf16* O = (__bf16*)C;
        long base = ((long)bb * cT + mt + hi2 * 8) * cQKV + hh * cD + nt + rlo;
#pragma unroll
        for (int g = 0; g < 8; ++g)
          O[base + (long)g * cQKV] = (__bf16)acc[i][j][g];
      } else if constexpr (EP == EP_F32_BIAS_RES) {
        float* O = (float*)C;
        const float bv = bias[nt + rlo];
        long base = (long)(mt + hi2 * 8) * ldc + nt + rlo;
#pragma unroll
        for (int g = 0; g < 8; ++g)
          O[base + (long)g * ldc] = acc[i][j][g] + bv + res[base + (long)g * ldc];
      } else {  // EP_BF16_BIAS_GELU (exact erf gelu)
        __bf16* O = (__bf16*)C;
        const float bv = bias[nt + rlo];
        long base = (long)(mt + hi2 * 8) * ldc + nt + rlo;
#pragma unroll
        for (int g = 0; g < 8; ++g) {
          float xg = acc[i][j][g] + bv;
          float gg = 0.5f * xg * (1.0f + erff(xg * 0.70710678118654752f));
          O[base + (long)g * ldc] = (__bf16)gg;
        }
      }
    }
  }
}

// ---------------------------------------------------------------------------
// LayerNorm: one 256-thread block per row of N (=768) f32, write bf16.
// ---------------------------------------------------------------------------
__global__ __launch_bounds__(256)
void layernorm_f32_bf16(const float* __restrict__ X,
                        const float* __restrict__ g, const float* __restrict__ b,
                        __bf16* __restrict__ O, int N)
{
  __shared__ float red[256];
  const long row = blockIdx.x;
  const float* x = X + row * N;
  const int tid = threadIdx.x;

  float s = 0.f, s2 = 0.f;
  for (int j = tid; j < N; j += 256) { float v = x[j]; s += v; s2 += v * v; }

  red[tid] = s; __syncthreads();
  for (int o = 128; o > 0; o >>= 1) { if (tid < o) red[tid] += red[tid + o]; __syncthreads(); }
  const float mean = red[0] / N; __syncthreads();

  red[tid] = s2; __syncthreads();
  for (int o = 128; o > 0; o >>= 1) { if (tid < o) red[tid] += red[tid + o]; __syncthreads(); }
  const float var = red[0] / N - mean * mean;
  const float r = rsqrtf(var + 1e-5f);

  for (int j = tid; j < N; j += 256)
    O[row * N + j] = (__bf16)((x[j] - mean) * r * g[j] + b[j]);
}

// ---------------------------------------------------------------------------
// Causal softmax: one block per (b,h,t) row of T=2048 f32 scores -> bf16 probs.
// Columns > t are written as 0 so P@V can run as a dense GEMM.
// ---------------------------------------------------------------------------
__global__ __launch_bounds__(256)
void softmax_causal(const float* __restrict__ S, __bf16* __restrict__ P)
{
  __shared__ float red[256];
  const long row = blockIdx.x;            // (b*H + h)*T + t
  const int  t   = (int)(row % cT);
  const int  L   = t + 1;
  const int  tid = threadIdx.x;
  const float* s = S + row * (long)cT;
  __bf16*      p = P + row * (long)cT;

  float mx = -3.4e38f;
  for (int j = tid; j < L; j += 256) mx = fmaxf(mx, s[j]);
  red[tid] = mx; __syncthreads();
  for (int o = 128; o > 0; o >>= 1) { if (tid < o) red[tid] = fmaxf(red[tid], red[tid + o]); __syncthreads(); }
  const float rmax = red[0]; __syncthreads();

  float sum = 0.f;
  for (int j = tid; j < L; j += 256) sum += __expf(s[j] - rmax);
  red[tid] = sum; __syncthreads();
  for (int o = 128; o > 0; o >>= 1) { if (tid < o) red[tid] += red[tid + o]; __syncthreads(); }
  const float inv = 1.0f / red[0];

  for (int j = tid; j < cT; j += 256)
    p[j] = (j < L) ? (__bf16)(__expf(s[j] - rmax) * inv) : (__bf16)0.0f;
}

// ---------------------------------------------------------------------------
// f32 -> bf16 convert (weights), 4 elements per thread.
// ---------------------------------------------------------------------------
__global__ __launch_bounds__(256)
void f32_to_bf16(const float* __restrict__ src, __bf16* __restrict__ dst, long n)
{
  long i = ((long)blockIdx.x * 256 + threadIdx.x) * 4;
#pragma unroll
  for (int k = 0; k < 4; ++k)
    if (i + k < n) dst[i + k] = (__bf16)src[i + k];
}

// ---------------------------------------------------------------------------
// Orchestration
// ---------------------------------------------------------------------------
extern "C" void kernel_launch(void* const* d_in, const int* in_sizes, int n_in,
                              void* d_out, int out_size, void* d_ws, size_t ws_size,
                              hipStream_t stream)
{
  (void)in_sizes; (void)n_in; (void)out_size; (void)ws_size;

  const float* x    = (const float*)d_in[0];
  const float* Wq   = (const float*)d_in[1];
  const float* Wk   = (const float*)d_in[2];
  const float* Wv   = (const float*)d_in[3];
  const float* Wo   = (const float*)d_in[4];
  const float* bo   = (const float*)d_in[5];
  const float* W1   = (const float*)d_in[6];
  const float* b1   = (const float*)d_in[7];
  const float* W2   = (const float*)d_in[8];
  const float* b2   = (const float*)d_in[9];
  const float* g1   = (const float*)d_in[10];
  const float* be1  = (const float*)d_in[11];
  const float* g2   = (const float*)d_in[12];
  const float* be2  = (const float*)d_in[13];

  const int M = 2 * cT;  // 4096 rows (B*T)

  // ---- workspace carve (256B aligned) ----
  size_t off = 0;
  auto carve = [&](size_t bytes) -> char* {
    char* p = (char*)d_ws + off;
    off += (bytes + 255) & ~(size_t)255;
    return p;
  };
  __bf16* hb  = (__bf16*)carve((size_t)M * cE * 2);            // LN1 output
  __bf16* WqB = (__bf16*)carve((size_t)cE * cQKV * 2);
  __bf16* WkB = (__bf16*)carve((size_t)cE * cQKV * 2);
  __bf16* WvB = (__bf16*)carve((size_t)cE * cQKV * 2);
  __bf16* WoB = (__bf16*)carve((size_t)cQKV * cE * 2);
  __bf16* W1B = (__bf16*)carve((size_t)cE * cFF * 2);
  __bf16* W2B = (__bf16*)carve((size_t)cFF * cE * 2);
  __bf16* Qb  = (__bf16*)carve((size_t)2 * cH * cT * cD * 2);  // [b,h,t,d]
  __bf16* KtB = (__bf16*)carve((size_t)2 * cH * cD * cT * 2);  // [b,h,d,t]
  __bf16* Vb  = (__bf16*)carve((size_t)2 * cH * cT * cD * 2);  // [b,h,t,d]
  float*  Sb  = (float*) carve((size_t)2 * cH * cT * cT * 4);  // scores f32
  __bf16* Pb  = (__bf16*)carve((size_t)2 * cH * cT * cT * 2);  // probs bf16
  __bf16* yc  = (__bf16*)carve((size_t)M * cQKV * 2);          // attn concat
  float*  x1  = (float*) carve((size_t)M * cE * 4);            // x + attn
  __bf16* h2  = (__bf16*)carve((size_t)M * cE * 2);            // LN2 output
  __bf16* ff1 = (__bf16*)carve((size_t)M * cFF * 2);           // gelu(h2@W1+b1)

  // ---- 1) weight fp32 -> bf16 ----
  auto cv = [&](const float* s, __bf16* d, long n) {
    int blocks = (int)((n + 256L * 4 - 1) / (256L * 4));
    f32_to_bf16<<<blocks, 256, 0, stream>>>(s, d, n);
  };
  cv(Wq, WqB, (long)cE * cQKV);
  cv(Wk, WkB, (long)cE * cQKV);
  cv(Wv, WvB, (long)cE * cQKV);
  cv(Wo, WoB, (long)cQKV * cE);
  cv(W1, W1B, (long)cE * cFF);
  cv(W2, W2B, (long)cFF * cE);

  // ---- 2) LN1 ----
  layernorm_f32_bf16<<<M, 256, 0, stream>>>(x, g1, be1, hb, cE);

  // ---- 3) QKV projections (M=4096, N=9216, K=768) ----
  dim3 gq(cQKV / BN, M / BM, 1);
  gemm_wmma_bf16<EP_QKV><<<gq, 256, 0, stream>>>(hb, WqB, Qb, M, cQKV, cE,
      0, 0, 0, 0, nullptr, nullptr, 1.f, 0);
  gemm_wmma_bf16<EP_KT ><<<gq, 256, 0, stream>>>(hb, WkB, KtB, M, cQKV, cE,
      0, 0, 0, 0, nullptr, nullptr, 1.f, 0);
  gemm_wmma_bf16<EP_QKV><<<gq, 256, 0, stream>>>(hb, WvB, Vb, M, cQKV, cE,
      0, 0, 0, 0, nullptr, nullptr, 1.f, 0);

  // ---- 4) scores S = (Q @ K^T) / 96, per (b,h): M=N=2048, K=768 ----
  dim3 gs(cT / BN, cT / BM, 2 * cH);
  gemm_wmma_bf16<EP_F32SCALE><<<gs, 256, 0, stream>>>(Qb, KtB, Sb, cT, cT, cD,
      (long)cT * cD, (long)cD * cT, (long)cT * cT, cT,
      nullptr, nullptr, 1.0f / 96.0f, /*causal=*/1);

  // ---- 5) causal softmax -> P (bf16, zero above diagonal) ----
  softmax_causal<<<2 * cH * cT, 256, 0, stream>>>(Sb, Pb);

  // ---- 6) y = P @ V, per (b,h): M=2048, N=768, K=2048; store to [b,t,h*D+d] ----
  dim3 gy(cD / BN, cT / BM, 2 * cH);
  gemm_wmma_bf16<EP_PV><<<gy, 256, 0, stream>>>(Pb, Vb, yc, cT, cD, cT,
      (long)cT * cT, (long)cT * cD, 0, 0, nullptr, nullptr, 1.f, 0);

  // ---- 7) x1 = x + yc @ Wo + bo  (M=4096, N=768, K=9216) ----
  dim3 go(cE / BN, M / BM, 1);
  gemm_wmma_bf16<EP_F32_BIAS_RES><<<go, 256, 0, stream>>>(yc, WoB, x1, M, cE, cQKV,
      0, 0, 0, cE, bo, x, 1.f, 0);

  // ---- 8) LN2 ----
  layernorm_f32_bf16<<<M, 256, 0, stream>>>(x1, g2, be2, h2, cE);

  // ---- 9) ff1 = gelu(h2 @ W1 + b1)  (M=4096, N=3072, K=768) ----
  dim3 gf1(cFF / BN, M / BM, 1);
  gemm_wmma_bf16<EP_BF16_BIAS_GELU><<<gf1, 256, 0, stream>>>(h2, W1B, ff1, M, cFF, cE,
      0, 0, 0, cFF, b1, nullptr, 1.f, 0);

  // ---- 10) out = x1 + ff1 @ W2 + b2  (M=4096, N=768, K=3072) ----
  dim3 gf2(cE / BN, M / BM, 1);
  gemm_wmma_bf16<EP_F32_BIAS_RES><<<gf2, 256, 0, stream>>>(ff1, W2B, (float*)d_out, M, cE, cFF,
      0, 0, 0, cE, b2, x1, 1.f, 0);
}
